// MultiheadPropagationLayer_71811853189803
// MI455X (gfx1250) — matrix operand order repaired and checked by reading, not compiled, not run on hardware
//
#include <hip/hip_runtime.h>
#include <hip/hip_bf16.h>
#include <math.h>

// ---------------------------------------------------------------------------
// GAT layer, fused flash-style. MI455X (gfx1250), wave32, WMMA f16->f32.
// adj stream (256 MB @ 23.3 TB/s ~= 11 us) is the roofline floor; all matrix
// work rides on v_wmma_f32_16x16x32_f16 with B fragments loaded as contiguous
// b128 segments from transposed f16 copies (no LDS in the hot loop). B
// fragments are materialized in bulk before each WMMA chain so loads clause
// together and WMMAs drain on staggered s_wait_loadcnt.
// ---------------------------------------------------------------------------

typedef _Float16 half8   __attribute__((ext_vector_type(8)));
typedef _Float16 half16  __attribute__((ext_vector_type(16)));
typedef float    float8  __attribute__((ext_vector_type(8)));

#define N_NODES 8192
#define F_IN    256
#define F_OUT   64
#define ALPHA_NS 0.2f
#define KEEP    0.5f

__device__ __forceinline__ float lrelu(float x) { return x > 0.f ? x : ALPHA_NS * x; }

// ------------------------- kernel 1: conversions ---------------------------
__global__ void cvt_kernel(const float* __restrict__ X, const float* __restrict__ W,
                           _Float16* __restrict__ X16, _Float16* __restrict__ W16T) {
  int tid = blockIdx.x * blockDim.x + threadIdx.x;
  int stride = gridDim.x * blockDim.x;
  for (int i = tid; i < N_NODES * F_IN; i += stride) X16[i] = (_Float16)X[i];
  for (int i = tid; i < F_IN * F_OUT; i += stride) {
    int k = i >> 6, c = i & 63;                 // W is [256][64]
    W16T[(size_t)c * F_IN + k] = (_Float16)W[i];
  }
}

// ------------------------- kernel 2: h = X @ W via WMMA --------------------
__global__ void __launch_bounds__(128) gemm_h_kernel(
    const _Float16* __restrict__ X16, const _Float16* __restrict__ W16T,
    float* __restrict__ h32) {
  const int t = threadIdx.x;
  const int wave = t >> 5, lane = t & 31;
  const int m = lane & 15, hi = lane >> 4;
  const int i0 = (blockIdx.x * 4 + wave) * 16;

  float8 c[4] = {};
  const _Float16* xrow = X16 + (size_t)(i0 + m) * F_IN + 8 * hi;
  const _Float16* wcol = W16T + (size_t)m * F_IN + 8 * hi;

  for (int k0 = 0; k0 < F_IN; k0 += 32) {
    // A fragment (16x32): lane m holds K {0..7,16..23} (hi=0) / {8..15,24..31} (hi=1)
    union { half16 v; half8 h2[2]; } a;
    a.h2[0] = *(const half8*)(xrow + k0);
    a.h2[1] = *(const half8*)(xrow + k0 + 16);
    // materialize all 4 B fragments first -> one load clause, staggered waits
    union { half16 v; half8 h2[2]; } b[4];
#pragma unroll
    for (int n = 0; n < 4; ++n) {
      const half8* p = (const half8*)(wcol + (size_t)n * 16 * F_IN + k0);
      b[n].h2[0] = p[0];
      b[n].h2[1] = p[2];
    }
#pragma unroll
    for (int n = 0; n < 4; ++n)
      c[n] = __builtin_amdgcn_wmma_f32_16x16x32_f16(false, a.v, false, b[n].v,
                                                    (short)0, c[n], false, false);
  }
#pragma unroll
  for (int n = 0; n < 4; ++n)
#pragma unroll
    for (int r = 0; r < 8; ++r)
      h32[(size_t)(i0 + r + 8 * hi) * F_OUT + n * 16 + m] = c[n][r];
}

// ------------------------- kernel 3: h16T = transpose(h) in f16 ------------
__global__ void __launch_bounds__(256) hT_kernel(const float* __restrict__ h32,
                                                 _Float16* __restrict__ h16T) {
  __shared__ _Float16 tile[64][65];
  const int j0 = blockIdx.x * 64;
  for (int lin = threadIdx.x; lin < 4096; lin += 256) {
    int r = lin >> 6, c = lin & 63;
    tile[c][r] = (_Float16)h32[(size_t)(j0 + r) * F_OUT + c];
  }
  __syncthreads();
  for (int lin = threadIdx.x; lin < 4096; lin += 256) {
    int c = lin >> 6, r = lin & 63;
    h16T[(size_t)c * N_NODES + j0 + r] = tile[c][r];
  }
}

// ------------------------- kernel 4: s1 = h@a1, s2 = h@a2 ------------------
__global__ void s12_kernel(const float* __restrict__ h32, const float* __restrict__ a,
                           float* __restrict__ s1, float* __restrict__ s2) {
  int i = blockIdx.x * blockDim.x + threadIdx.x;
  if (i >= N_NODES) return;
  const float* hr = h32 + (size_t)i * F_OUT;
  float v1 = 0.f, v2 = 0.f;
#pragma unroll
  for (int k = 0; k < F_OUT; ++k) { v1 += hr[k] * a[k]; v2 += hr[k] * a[F_OUT + k]; }
  s1[i] = v1; s2[i] = v2;
}

// ------------------------- kernel 5: max(s2) -------------------------------
__global__ void s2max_kernel(const float* __restrict__ s2, float* __restrict__ out) {
  __shared__ float red[256];
  float mx = -3.0e38f;
  for (int i = threadIdx.x; i < N_NODES; i += 256) mx = fmaxf(mx, s2[i]);
  red[threadIdx.x] = mx; __syncthreads();
  for (int s = 128; s > 0; s >>= 1) {
    if (threadIdx.x < s) red[threadIdx.x] = fmaxf(red[threadIdx.x], red[threadIdx.x + s]);
    __syncthreads();
  }
  if (threadIdx.x == 0) out[0] = red[0];
}

// ------------------------- kernel 6: fused masked softmax + att@h ----------
// One block per 16-row tile (512 blocks), 4 waves; wave w covers j in
// [w*2048,(w+1)*2048). m_i = lrelu(s1[i]+max(s2)) is a softmax upper bound,
// so weights are exp(e-m_i) with no online rescaling -> pure WMMA accumulate.
__global__ void __launch_bounds__(128) attn_kernel(
    const int* __restrict__ adj, const float* __restrict__ h32,
    const _Float16* __restrict__ h16T, const float* __restrict__ s1,
    const float* __restrict__ s2, const float* __restrict__ s2max,
    float* __restrict__ out) {
  __shared__ float sAcc[4][16][F_OUT];        // 16 KB (epilogue reduce only)
  __shared__ float sSum[4][16];
  __shared__ float sCnt[4][16];

  const int t = threadIdx.x, wave = t >> 5, lane = t & 31;
  const int m = lane & 15, hi = lane >> 4;
  const int i0 = blockIdx.x * 16;

  const float s1m  = s1[i0 + m];
  const float mrow = lrelu(s1m + s2max[0]);   // >= max_j lrelu(s1m + s2[j])

  const int jbase = wave * (N_NODES / 4);
  const int*      adjRow = adj + (size_t)(i0 + m) * N_NODES + 8 * hi;
  const float*    s2p    = s2 + 8 * hi;
  const _Float16* bcol   = h16T + (size_t)m * N_NODES + 8 * hi;

  float8 c[4] = {};
  float sum_part = 0.f, cnt_part = 0.f;

  for (int step = 0; step < 64; ++step) {
    const int j0 = jbase + step * 32;

    // ---- issue all loads for this chunk up front (adj, s2, 4 B fragments) --
    const int* ap   = adjRow + j0;
    const float* sp = s2p + j0;
    int4  a0 = *(const int4*)(ap);       int4  a1 = *(const int4*)(ap + 4);
    int4  a2 = *(const int4*)(ap + 16);  int4  a3 = *(const int4*)(ap + 20);
    float4 v0 = *(const float4*)(sp);      float4 v1 = *(const float4*)(sp + 4);
    float4 v2 = *(const float4*)(sp + 16); float4 v3 = *(const float4*)(sp + 20);

    union { half16 v; half8 h2[2]; } b[4];
#pragma unroll
    for (int n = 0; n < 4; ++n) {
      const half8* p = (const half8*)(bcol + (size_t)n * 16 * N_NODES + j0);
      b[n].h2[0] = p[0];
      b[n].h2[1] = p[2];
    }
    __builtin_prefetch((const void*)(ap + 32), 0, 1);  // next adj chunk

    // ---- A fragment: softmax weights for rows i0..i0+15, cols j0..j0+31 ----
    int   av[16] = {a0.x,a0.y,a0.z,a0.w, a1.x,a1.y,a1.z,a1.w,
                    a2.x,a2.y,a2.z,a2.w, a3.x,a3.y,a3.z,a3.w};
    float sv[16] = {v0.x,v0.y,v0.z,v0.w, v1.x,v1.y,v1.z,v1.w,
                    v2.x,v2.y,v2.z,v2.w, v3.x,v3.y,v3.z,v3.w};

    union { half16 v; _Float16 e[16]; } a;
#pragma unroll
    for (int q = 0; q < 16; ++q) {
      float e  = lrelu(s1m + sv[q]);
      bool  ok = av[q] > 0;
      float w  = ok ? __expf(e - mrow) : 0.f;   // masked entries contribute 0
      sum_part += w;
      cnt_part += ok ? 1.f : 0.f;
      a.e[q] = (_Float16)w;
    }

    // ---- 4 WMMAs: acc(16x64) += weights(16x32) @ h(32x64) ----
#pragma unroll
    for (int n = 0; n < 4; ++n)
      c[n] = __builtin_amdgcn_wmma_f32_16x16x32_f16(false, a.v, false, b[n].v,
                                                    (short)0, c[n], false, false);
  }

  // combine the two K-halves of each row, publish per-wave partials
  float sum_all = sum_part + __shfl_xor(sum_part, 16, 32);
  float cnt_all = cnt_part + __shfl_xor(cnt_part, 16, 32);
  if (lane < 16) { sSum[wave][m] = sum_all; sCnt[wave][m] = cnt_all; }
#pragma unroll
  for (int n = 0; n < 4; ++n)
#pragma unroll
    for (int r = 0; r < 8; ++r)
      sAcc[wave][r + 8 * hi][n * 16 + m] = c[n][r];
  __syncthreads();

  // cross-wave reduce + epilogue: mask fallback, 0.5 blend, ELU
  for (int q = 0; q < 8; ++q) {
    int lin = t + q * 128;
    int row = lin >> 6, col = lin & 63;
    float acc = sAcc[0][row][col] + sAcc[1][row][col] +
                sAcc[2][row][col] + sAcc[3][row][col];
    float tot = sSum[0][row] + sSum[1][row] + sSum[2][row] + sSum[3][row];
    float cnt = sCnt[0][row] + sCnt[1][row] + sCnt[2][row] + sCnt[3][row];
    float hv  = h32[(size_t)(i0 + row) * F_OUT + col];
    float hp  = tot > 0.f ? acc / tot : 0.f;
    hp = (cnt > 0.f) ? hp : hv;                   // zero-degree rows keep h
    float v = (1.f - KEEP) * hp + KEEP * hv;      // KEEP_RATIO blend
    out[(size_t)(i0 + row) * F_OUT + col] = v > 0.f ? v : __expf(v) - 1.f;  // ELU
  }
}

// ---------------------------------------------------------------------------
extern "C" void kernel_launch(void* const* d_in, const int* in_sizes, int n_in,
                              void* d_out, int out_size, void* d_ws, size_t ws_size,
                              hipStream_t stream) {
  const float* X   = (const float*)d_in[0];   // 8192 x 256
  const int*   adj = (const int*)d_in[1];     // 8192 x 8192
  const float* Wm  = (const float*)d_in[2];   // 256 x 64
  const float* av  = (const float*)d_in[3];   // 128
  float* out = (float*)d_out;                 // 8192 x 64

  // workspace carve-up (~7.2 MB)
  char* ws = (char*)d_ws;
  size_t off = 0;
  auto carve = [&](size_t bytes) { char* p = ws + off; off = (off + bytes + 255) & ~(size_t)255; return p; };
  _Float16* X16  = (_Float16*)carve((size_t)N_NODES * F_IN * 2);
  _Float16* W16T = (_Float16*)carve((size_t)F_IN * F_OUT * 2);
  float*    h32  = (float*)   carve((size_t)N_NODES * F_OUT * 4);
  _Float16* h16T = (_Float16*)carve((size_t)N_NODES * F_OUT * 2);
  float*    s1   = (float*)   carve((size_t)N_NODES * 4);
  float*    s2   = (float*)   carve((size_t)N_NODES * 4);
  float*    s2m  = (float*)   carve(256);
  (void)ws_size; (void)in_sizes; (void)n_in; (void)out_size;

  cvt_kernel   <<<2048, 256, 0, stream>>>(X, Wm, X16, W16T);
  gemm_h_kernel<<<128, 128, 0, stream>>>(X16, W16T, h32);
  hT_kernel    <<<N_NODES / 64, 256, 0, stream>>>(h32, h16T);
  s12_kernel   <<<N_NODES / 256, 256, 0, stream>>>(h32, av, s1, s2);
  s2max_kernel <<<1, 256, 0, stream>>>(s2, s2m);
  attn_kernel  <<<N_NODES / 16, 128, 0, stream>>>(adj, h32, h16T, s1, s2, s2m, out);
}
